// Net_40793599377677
// MI455X (gfx1250) — compile-verified, matrix operand order checked by reading, not run on hardware
//
#include <hip/hip_runtime.h>

typedef __attribute__((ext_vector_type(16))) __bf16 v16bf;
typedef __attribute__((ext_vector_type(8)))  float  v8f;
typedef __attribute__((ext_vector_type(4)))  float  f32x4;
typedef __attribute__((ext_vector_type(4)))  unsigned int u32x4;
typedef __attribute__((ext_vector_type(8)))  int    i32x8;
typedef __attribute__((ext_vector_type(4)))  int    i32x4;

#define NN 50000
#define NE 800000
#define NP 100000
#define DIN 256
#define DH  128

union Frag16 { v16bf v; f32x4 f[2]; };

__device__ __forceinline__ unsigned short f2bf(float f) {
    union { float f; unsigned int u; } c; c.f = f;
    unsigned int r = c.u + 0x7FFFu + ((c.u >> 16) & 1u);
    return (unsigned short)(r >> 16);
}

// ---- Tensor Data Mover: 2D bf16 tile from global -> LDS (6-arg builtin) ----
__device__ __forceinline__ void tdm_load_2d_bf16(const unsigned short* gsrc, unsigned lds_off,
                                                 unsigned tensor_d0, unsigned tensor_d1,
                                                 unsigned tile_d0, unsigned tile_d1,
                                                 unsigned row_stride)
{
    unsigned long long ga = (unsigned long long)(uintptr_t)gsrc;
    u32x4 g0;
    g0.x = 1u;                                   // count=1 (valid user descriptor)
    g0.y = lds_off;                              // LDS byte address
    g0.z = (unsigned)ga;                         // global_addr[31:0]
    g0.w = (unsigned)(ga >> 32) | (2u << 30);    // global_addr[56:32] | type=2 ("image")
    i32x8 g1;
    g1[0] = (int)(1u << 16);                     // workgroup_mask=0, data_size=1 (2 bytes)
    g1[1] = (int)((tensor_d0 & 0xFFFFu) << 16);  // tensor_dim0[15:0]
    g1[2] = (int)((tensor_d0 >> 16) | ((tensor_d1 & 0xFFFFu) << 16));
    g1[3] = (int)((tensor_d1 >> 16) | (tile_d0 << 16));
    g1[4] = (int)tile_d1;                        // tile_dim1 | tile_dim2=0
    g1[5] = (int)row_stride;                     // tensor_dim0_stride[31:0]
    g1[6] = 0;
    g1[7] = 0;
    i32x4 z4 = {0, 0, 0, 0};
    i32x8 z8 = {0, 0, 0, 0, 0, 0, 0, 0};
    __builtin_amdgcn_tensor_load_to_lds(g0, g1, z4, z4, z8, 0);
}

// ---- pack a [K x N(=128)] row-major f32 weight into bf16 WMMA B-fragment order ----
__global__ void pack_w(const float* __restrict__ W, unsigned short* __restrict__ out,
                       int K, int N) {
    int id = blockIdx.x * blockDim.x + threadIdx.x;
    if (id >= K * N) return;
    int numKT = K >> 5;
    int t    = id & 15;
    int lane = (id >> 4) & 31;
    int frag = id >> 9;
    int kt = frag % numKT;
    int nt = frag / numKT;
    int kl = (t < 8 ? t : t + 8) + ((lane >= 16) ? 8 : 0);
    int col = (nt << 4) + (lane & 15);
    out[id] = f2bf(W[(size_t)((kt << 5) + kl) * N + col]);
}

__global__ void f32_to_bf16(const float* __restrict__ in, unsigned short* __restrict__ out, int n) {
    int i = blockIdx.x * blockDim.x + threadIdx.x;
    if (i < n) out[i] = f2bf(in[i]);
}

__device__ __forceinline__ void loadB4(Frag16* dst, const unsigned short* __restrict__ Bp,
                                       int ntbase, int kt, int numKT, int lane) {
#pragma unroll
    for (int i = 0; i < 4; ++i) {
        const unsigned short* bp =
            Bp + (((size_t)((ntbase + i) * numKT + kt) * 32 + lane) << 4);
        dst[i].f[0] = *(const f32x4*)(bp);
        dst[i].f[1] = *(const f32x4*)(bp + 8);
    }
}

// ---- fused 4-projection GEMM: out_p = A(bf16 MxK) @ W_p(K x 128) + b_p, p = blockIdx.y ----
// TDM stages the 128xK A-tile into LDS; A-fragments via ds_load_b128; B fragments in a
// copy-free register ping-pong (K-loop unrolled by 2; numKT is always even here).
__global__ __launch_bounds__(256)
void gemm_qkvs(const unsigned short* __restrict__ A, const unsigned short* __restrict__ Bpk,
               const float* __restrict__ b0, const float* __restrict__ b1,
               const float* __restrict__ b2, const float* __restrict__ b3,
               float* __restrict__ o0, float* __restrict__ o1,
               float* __restrict__ o2, float* __restrict__ o3,
               int M, int K)
{
    extern __shared__ unsigned short As[];       // 128 * K bf16
    const int N = 128;
    const int numKT = K >> 5;
    const int proj = blockIdx.y;
    const float* bias = (proj == 0) ? b0 : (proj == 1) ? b1 : (proj == 2) ? b2 : b3;
    float* out        = (proj == 0) ? o0 : (proj == 1) ? o1 : (proj == 2) ? o2 : o3;
    const unsigned short* Bp = Bpk + (size_t)proj * K * N;

    const int wave = threadIdx.x >> 5;
    const int lane = threadIdx.x & 31;
    const int mblk = blockIdx.x * 128;
    const int m0 = mblk + (wave << 4);
    const bool active = (m0 < M);

    // Wave 0 DMAs the whole 128xK activation tile into LDS (TDM; per-wave issue).
    if (threadIdx.x < 32) {
        tdm_load_2d_bf16(A + (size_t)mblk * K, (unsigned)(uintptr_t)As,
                         (unsigned)K, (unsigned)(M - mblk),
                         (unsigned)K, 128u, (unsigned)K);
        __builtin_amdgcn_s_wait_tensorcnt(0);
    }
    __syncthreads();

    if (active) {
        const int rloc = (wave << 4) + (lane & 15);      // row within LDS tile
        const int kshift = (lane >= 16) ? 8 : 0;         // ISA 16-bit A striping
        const unsigned short* abase = As + (size_t)rloc * K + kshift;
        const int colb = lane & 15;
        const int rb = m0 + ((lane >= 16) ? 8 : 0);

#pragma unroll
        for (int h = 0; h < 2; ++h) {                    // two N-halves of 4 tiles
            const int ntb = h * 4;
            v8f acc[4];
#pragma unroll
            for (int i = 0; i < 4; ++i) acc[i] = (v8f){};

            Frag16 fA[4], fB[4];
            loadB4(fA, Bp, ntb, 0, numKT, lane);

            for (int kt = 0; kt < numKT; kt += 2) {      // numKT even: no odd tail
                Frag16 a;
                const unsigned short* ap = abase + (kt << 5);
                a.f[0] = *(const f32x4*)(ap);            // ds_load_b128
                a.f[1] = *(const f32x4*)(ap + 16);
                loadB4(fB, Bp, ntb, kt + 1, numKT, lane);
#pragma unroll
                for (int i = 0; i < 4; ++i)
                    acc[i] = __builtin_amdgcn_wmma_f32_16x16x32_bf16(
                        false, a.v, false, fA[i].v, (short)0, acc[i], false, false);

                const unsigned short* ap1 = abase + ((kt + 1) << 5);
                a.f[0] = *(const f32x4*)(ap1);
                a.f[1] = *(const f32x4*)(ap1 + 16);
                if (kt + 2 < numKT) loadB4(fA, Bp, ntb, kt + 2, numKT, lane);
#pragma unroll
                for (int i = 0; i < 4; ++i)
                    acc[i] = __builtin_amdgcn_wmma_f32_16x16x32_bf16(
                        false, a.v, false, fB[i].v, (short)0, acc[i], false, false);
            }

#pragma unroll
            for (int i = 0; i < 4; ++i) {
                const int col = ((ntb + i) << 4) + colb;
                const float bb = bias[col];
#pragma unroll
                for (int j = 0; j < 8; ++j) {
                    const int row = rb + j;
                    if (row < M) out[(size_t)row * N + col] = acc[i][j] + bb;
                }
            }
        }
    }
}

// ---- edge attention ----
__global__ void init_seg(float* __restrict__ amax, float* __restrict__ den, int n) {
    int i = blockIdx.x * blockDim.x + threadIdx.x;
    if (i < n) { amax[i] = -3.402823466e38f; den[i] = 0.0f; }
}

__device__ __forceinline__ void atomicMaxF(float* addr, float v) {
    if (v >= 0.0f) atomicMax((int*)addr, __float_as_int(v));
    else atomicMin((unsigned int*)addr, (unsigned int)__float_as_int(v));
}

__global__ __launch_bounds__(256)
void edge_alpha(const int* __restrict__ ei, const float* __restrict__ q,
                const float* __restrict__ k, float* __restrict__ alpha,
                float* __restrict__ amax, int E)
{
    int wid = (blockIdx.x * blockDim.x + threadIdx.x) >> 5;
    int lane = threadIdx.x & 31;
    if (wid >= E) return;
    int s = ei[wid], d = ei[E + wid];
    f32x4 qv = *(const f32x4*)(q + (size_t)d * DH + lane * 4);
    f32x4 kv = *(const f32x4*)(k + (size_t)s * DH + lane * 4);
    float p = qv.x * kv.x + qv.y * kv.y + qv.z * kv.z + qv.w * kv.w;
#pragma unroll
    for (int off = 16; off > 0; off >>= 1) p += __shfl_xor(p, off, 32);
    if (lane == 0) {
        float a = p * 0.0883883476483184f; // 1/sqrt(128)
        alpha[wid] = a;
        atomicMaxF(&amax[d], a);
    }
}

__global__ void edge_ex(const int* __restrict__ ei, float* __restrict__ alpha,
                        const float* __restrict__ amax, float* __restrict__ den, int E) {
    int e = blockIdx.x * blockDim.x + threadIdx.x;
    if (e >= E) return;
    int d = ei[E + e];
    float ex = __expf(alpha[e] - amax[d]);
    alpha[e] = ex;
    atomicAdd(&den[d], ex);
}

__global__ __launch_bounds__(256)
void edge_agg(const int* __restrict__ ei, const float* __restrict__ v,
              const float* __restrict__ alpha, const float* __restrict__ den,
              float* __restrict__ agg, int E)
{
    int wid = (blockIdx.x * blockDim.x + threadIdx.x) >> 5;
    int lane = threadIdx.x & 31;
    if (wid >= E) return;
    int s = ei[wid], d = ei[E + wid];
    float w = alpha[wid] / den[d];
    f32x4 vv = *(const f32x4*)(v + (size_t)s * DH + lane * 4);
    float* dst = agg + (size_t)d * DH + lane * 4;
    atomicAdd(dst + 0, vv.x * w);
    atomicAdd(dst + 1, vv.y * w);
    atomicAdd(dst + 2, vv.z * w);
    atomicAdd(dst + 3, vv.w * w);
}

__global__ void relu_k(float* __restrict__ x, int n) {
    int i = blockIdx.x * blockDim.x + threadIdx.x;
    if (i < n) x[i] = fmaxf(x[i], 0.0f);
}

__global__ __launch_bounds__(256)
void predict_k(const float* __restrict__ h, const int* __restrict__ pos,
               const int* __restrict__ neg, const float* __restrict__ Wl,
               const float* __restrict__ bl, float* __restrict__ out, int npred)
{
    int wid = (blockIdx.x * blockDim.x + threadIdx.x) >> 5;
    int lane = threadIdx.x & 31;
    if (wid >= 2 * npred) return;
    int a, b;
    if (wid < npred) { a = pos[wid]; b = pos[npred + wid]; }
    else { int j = wid - npred; a = neg[j]; b = neg[npred + j]; }
    f32x4 ha = *(const f32x4*)(h + (size_t)a * DH + lane * 4);
    f32x4 hb = *(const f32x4*)(h + (size_t)b * DH + lane * 4);
    f32x4 w  = *(const f32x4*)(Wl + lane * 4);
    float p = ha.x * hb.x * w.x + ha.y * hb.y * w.y + ha.z * hb.z * w.z + ha.w * hb.w * w.w;
#pragma unroll
    for (int off = 16; off > 0; off >>= 1) p += __shfl_xor(p, off, 32);
    if (lane == 0) out[wid] = p + bl[0];
}

extern "C" void kernel_launch(void* const* d_in, const int* in_sizes, int n_in,
                              void* d_out, int out_size, void* d_ws, size_t ws_size,
                              hipStream_t stream) {
    (void)in_sizes; (void)n_in; (void)out_size; (void)ws_size;
    const float* x   = (const float*)d_in[0];
    const int*   ei  = (const int*)d_in[1];
    const int*   pos = (const int*)d_in[2];
    const int*   neg = (const int*)d_in[3];
    const float* W1[4] = { (const float*)d_in[4], (const float*)d_in[6],
                           (const float*)d_in[8], (const float*)d_in[10] };
    const float* B1[4] = { (const float*)d_in[5], (const float*)d_in[7],
                           (const float*)d_in[9], (const float*)d_in[11] };
    const float* W2[4] = { (const float*)d_in[12], (const float*)d_in[14],
                           (const float*)d_in[16], (const float*)d_in[18] };
    const float* B2[4] = { (const float*)d_in[13], (const float*)d_in[15],
                           (const float*)d_in[17], (const float*)d_in[19] };
    const float* Wl = (const float*)d_in[20];
    const float* bl = (const float*)d_in[21];

    char* ws = (char*)d_ws;
    size_t off = 0;
    auto alloc = [&](size_t bytes) -> char* {
        char* p = ws + off; off += (bytes + 255) & ~(size_t)255; return p;
    };
    unsigned short* xb   = (unsigned short*)alloc((size_t)NN * DIN * 2);
    unsigned short* wpk1 = (unsigned short*)alloc((size_t)4 * DIN * DH * 2);
    unsigned short* wpk2 = (unsigned short*)alloc((size_t)4 * DH * DH * 2);
    float* q  = (float*)alloc((size_t)NN * DH * 4);
    float* k  = (float*)alloc((size_t)NN * DH * 4);
    float* v  = (float*)alloc((size_t)NN * DH * 4);
    float* s  = (float*)alloc((size_t)NN * DH * 4);   // skip target -> agg -> h
    float* al = (float*)alloc((size_t)NE * 4);
    float* am = (float*)alloc((size_t)NN * 4);
    float* dn = (float*)alloc((size_t)NN * 4);
    unsigned short* hb = (unsigned short*)alloc((size_t)NN * DH * 2);

    // pack weights into WMMA fragment order
    for (int p = 0; p < 4; ++p)
        pack_w<<<(DIN * DH + 255) / 256, 256, 0, stream>>>(W1[p], wpk1 + (size_t)p * DIN * DH, DIN, DH);
    for (int p = 0; p < 4; ++p)
        pack_w<<<(DH * DH + 255) / 256, 256, 0, stream>>>(W2[p], wpk2 + (size_t)p * DH * DH, DH, DH);

    // x -> bf16
    f32_to_bf16<<<((size_t)NN * DIN + 255) / 256, 256, 0, stream>>>(x, xb, NN * DIN);

    // layer 1 projections q,k,v,s  (LDS tile = 128 x 256 bf16 = 64KB)
    {
        dim3 g((NN + 127) / 128, 4);
        gemm_qkvs<<<g, 256, 128 * DIN * 2, stream>>>(xb, wpk1, B1[0], B1[1], B1[2], B1[3],
                                                     q, k, v, s, NN, DIN);
    }
    // layer 1 attention (softmax over incoming edges of dst), agg += into s (skip folded)
    init_seg<<<(NN + 255) / 256, 256, 0, stream>>>(am, dn, NN);
    edge_alpha<<<NE / 8, 256, 0, stream>>>(ei, q, k, al, am, NE);
    edge_ex<<<(NE + 255) / 256, 256, 0, stream>>>(ei, al, am, dn, NE);
    edge_agg<<<NE / 8, 256, 0, stream>>>(ei, v, al, dn, s, NE);
    relu_k<<<((size_t)NN * DH + 255) / 256, 256, 0, stream>>>(s, NN * DH);

    // h1 -> bf16, layer 2 projections (reuse q,k,v; s overwritten after hb exists)
    f32_to_bf16<<<((size_t)NN * DH + 255) / 256, 256, 0, stream>>>(s, hb, NN * DH);
    {
        dim3 g((NN + 127) / 128, 4);
        gemm_qkvs<<<g, 256, 128 * DH * 2, stream>>>(hb, wpk2, B2[0], B2[1], B2[2], B2[3],
                                                    q, k, v, s, NN, DH);
    }
    init_seg<<<(NN + 255) / 256, 256, 0, stream>>>(am, dn, NN);
    edge_alpha<<<NE / 8, 256, 0, stream>>>(ei, q, k, al, am, NE);
    edge_ex<<<(NE + 255) / 256, 256, 0, stream>>>(ei, al, am, dn, NE);
    edge_agg<<<NE / 8, 256, 0, stream>>>(ei, v, al, dn, s, NE);

    // final edge scorer over pos++neg
    predict_k<<<(2 * NP) / 8, 256, 0, stream>>>(s, pos, neg, Wl, bl, (float*)d_out, NP);
}
